// CausalSelfAttention_70291434766691
// MI455X (gfx1250) — matrix operand order repaired and checked by reading, not compiled, not run on hardware
//
#include <hip/hip_runtime.h>
#include <hip/hip_bf16.h>

// ---------------- problem constants ----------------
constexpr int BATCH = 2;
constexpr int SEQ   = 2048;
constexpr int EMB   = 2048;
constexpr int NH    = 16;
constexpr int NKVH  = 4;
constexpr int HDIM  = 128;            // EMB / NH
constexpr int KVD   = NKVH * HDIM;    // 512
constexpr int MR    = BATCH * SEQ;    // 4096 rows for all projections

typedef __bf16 bf16;
typedef __attribute__((ext_vector_type(16))) __bf16 v16bf;
typedef __attribute__((ext_vector_type(8)))  float  v8f;

// ---------------- CDNA5 async copy helpers ----------------
// GLOBAL_LOAD_ASYNC_TO_LDS_B128 (GV mode): each active lane moves 16B from its
// 64-bit global address into LDS at its 32-bit LDS byte address (ASYNCcnt).
__device__ __forceinline__ void async_load_b128(uint32_t lds_off, const void* gptr) {
  asm volatile("global_load_async_to_lds_b128 %0, %1, off"
               :: "v"(lds_off), "v"((uint64_t)(uintptr_t)gptr)
               : "memory");
}
__device__ __forceinline__ void wait_async() {
  asm volatile("s_wait_asynccnt 0x0" ::: "memory");
}
// LDS byte address = low 32 bits of the generic pointer (ISA 10.2 aperture rule)
__device__ __forceinline__ uint32_t lds_addr(const void* p) {
  return (uint32_t)(uintptr_t)p;
}

// ---------------- pure-VALU cross-lane reductions (DPP16, no LDS) ----------
template <int CTRL>
__device__ __forceinline__ float dpp_mov_f(float x) {
  return __int_as_float(__builtin_amdgcn_update_dpp(
      0, __float_as_int(x), CTRL, 0xF, 0xF, true));
}
// max over each 16-lane half: xor1, xor2, half-mirror, mirror
__device__ __forceinline__ float half_max16(float x) {
  x = fmaxf(x, dpp_mov_f<0xB1>(x));    // quad_perm [1,0,3,2]
  x = fmaxf(x, dpp_mov_f<0x4E>(x));    // quad_perm [2,3,0,1]
  x = fmaxf(x, dpp_mov_f<0x141>(x));   // row_half_mirror
  x = fmaxf(x, dpp_mov_f<0x140>(x));   // row_mirror
  return x;
}
// sum over all 32 lanes: DPP within halves + SWAPX16 swizzle across halves
__device__ __forceinline__ float wave_sum32(float x) {
  x += dpp_mov_f<0xB1>(x);
  x += dpp_mov_f<0x4E>(x);
  x += dpp_mov_f<0x141>(x);
  x += dpp_mov_f<0x140>(x);
  x += __int_as_float(__builtin_amdgcn_ds_swizzle(__float_as_int(x), 0x401F));
  return x;
}

// ---------------- fp32 -> bf16 pack ----------------
__global__ void cvt_f32_bf16(const float* __restrict__ in, bf16* __restrict__ out, int n4) {
  int i = blockIdx.x * blockDim.x + threadIdx.x;
  if (i >= n4) return;
  float4 f = ((const float4*)in)[i];
  union { bf16 h[4]; uint2 u; } r;
  r.h[0] = (bf16)f.x; r.h[1] = (bf16)f.y; r.h[2] = (bf16)f.z; r.h[3] = (bf16)f.w;
  ((uint2*)out)[i] = r.u;
}

// ---------------- WMMA GEMM: C[M,N] = A[M,K] * W[N,K]^T ----------------
// Block: 256 thr (8 waves). Block tile 128x128, wave tile 32x64 (2x4 C tiles):
// 12 LDS b128 loads per 8 WMMAs. A/B tiles double-buffered in LDS, filled with
// async global->LDS copies that overlap the WMMA work on the other buffer.
// MODE 0: C row-major (M,N).  MODE 1: (b, head, s, e).  MODE 2: (b, head, e, s)
// (transposed heads — used for V so attention reads V^T contiguously).
template<typename OutT, int MODE>
__global__ void __launch_bounds__(256)
gemm_bf16_wmma(const bf16* __restrict__ A, const bf16* __restrict__ W,
               OutT* __restrict__ C, int M, int N, int K) {
  // rows padded to 40 elems (80B): 20-bank row stride -> conflict-free
  __shared__ __align__(16) bf16 As[2][128][40];
  __shared__ __align__(16) bf16 Bs[2][128][40];
  const int t    = threadIdx.x;
  const int lane = t & 31, wave = t >> 5;
  const int ln   = lane & 15, lh = lane >> 4;
  const int waveN = wave & 1, waveM = wave >> 1;     // 4 M-groups x 2 N-groups
  const int rowBase = blockIdx.y * 128;
  const int colBase = blockIdx.x * 128;

  auto stage = [&](int buf, int k0) {
    // A and B tiles: 128 rows x 64B each, two b128 per thread per tile
#pragma unroll
    for (int r = 0; r < 2; ++r) {
      const int c = r * 256 + t;
      const int row = c >> 2, cb = (c & 3) * 16;
      async_load_b128(lds_addr(&As[buf][row][0]) + cb,
                      (const char*)(A + (size_t)(rowBase + row) * K + k0) + cb);
      async_load_b128(lds_addr(&Bs[buf][row][0]) + cb,
                      (const char*)(W + (size_t)(colBase + row) * K + k0) + cb);
    }
  };

  v8f acc[2][4] = {};

  stage(0, 0);
  wait_async();
  __syncthreads();

  for (int k0 = 0; k0 < K; k0 += 32) {
    const int cur = (k0 >> 5) & 1;
    if (k0 + 32 < K) stage(cur ^ 1, k0 + 32);   // overlap copy with compute

    // two A fragments (16x32): lane-half holds K 0..15 / 16..31
    union { v16bf v; uint4 u[2]; } au[2];
#pragma unroll
    for (int a = 0; a < 2; ++a) {
      const bf16* arow = &As[cur][waveM * 32 + a * 16 + ln][0];
      au[a].u[0] = *(const uint4*)(arow + 16 * lh);
      au[a].u[1] = *(const uint4*)(arow + 16 * lh + 8);
    }

#pragma unroll
    for (int c = 0; c < 4; ++c) {
      // B fragment (32x16): lane = n, lane-half selects K 0..15 / 16..31
      union { v16bf v; uint4 u[2]; } bu;
      const bf16* brow = &Bs[cur][waveN * 64 + c * 16 + ln][0];
      bu.u[0] = *(const uint4*)(brow + 16 * lh);
      bu.u[1] = *(const uint4*)(brow + 16 * lh + 8);
#pragma unroll
      for (int a = 0; a < 2; ++a)
        acc[a][c] = __builtin_amdgcn_wmma_f32_16x16x32_bf16(
            false, au[a].v, false, bu.v, (short)0, acc[a][c], false, false);
    }

    wait_async();      // next tile landed (had whole compute phase to fly)
    __syncthreads();
  }

#pragma unroll
  for (int a = 0; a < 2; ++a)
#pragma unroll
    for (int c = 0; c < 4; ++c)
#pragma unroll
      for (int i = 0; i < 8; ++i) {
        const int m = rowBase + waveM * 32 + a * 16 + i + 8 * lh;
        const int n = colBase + waveN * 64 + c * 16 + ln;
        const float val = acc[a][c][i];
        size_t off;
        if (MODE == 1) {
          const int bI = m / SEQ, sI = m % SEQ;
          const int head = n / HDIM, e = n % HDIM;
          off = (((size_t)bI * (N / HDIM) + head) * SEQ + sI) * HDIM + e;
        } else if (MODE == 2) {
          const int bI = m / SEQ, sI = m % SEQ;
          const int head = n / HDIM, e = n % HDIM;
          off = (((size_t)bI * (N / HDIM) + head) * HDIM + e) * SEQ + sI;
        } else {
          off = (size_t)m * N + n;
        }
        C[off] = (OutT)val;
      }
}

// ---------------- RMSNorm + RoPE (+ optional per-head gain), in place -------
// One wave per (b, head, s) row of 128. Lane l owns elems {l, l+32, l+64, l+96}.
__global__ void __launch_bounds__(256)
rmsnorm_rope(bf16* __restrict__ t, const float* __restrict__ gain,
             int rows, int heads) {
  const int w    = (blockIdx.x * blockDim.x + threadIdx.x) >> 5;
  const int lane = threadIdx.x & 31;
  if (w >= rows) return;
  const int s = w % SEQ;
  const int h = (w / SEQ) % heads;
  bf16* row = t + (size_t)w * HDIM;

  float f0 = (float)row[lane];
  float f1 = (float)row[lane + 32];
  float f2 = (float)row[lane + 64];
  float f3 = (float)row[lane + 96];
  float ss = wave_sum32(f0 * f0 + f1 * f1 + f2 * f2 + f3 * f3);
  float r = rsqrtf(ss * (1.0f / 128.0f) + 1.1920929e-7f);
  if (gain) r *= gain[h];           // gain commutes with the rotation
  f0 *= r; f1 *= r; f2 *= r; f3 *= r;

  const float L2B = 13.287712379549449f;  // log2(10000)
  const float fra = (float)s * exp2f(-(float)lane        * (L2B / 64.0f));
  const float frb = (float)s * exp2f(-(float)(lane + 32) * (L2B / 64.0f));
  float ca, sa, cb, sb;
  __sincosf(fra, &sa, &ca);
  __sincosf(frb, &sb, &cb);
  row[lane]      = (bf16)( f0 * ca + f2 * sa);
  row[lane + 64] = (bf16)(-f0 * sa + f2 * ca);
  row[lane + 32] = (bf16)( f1 * cb + f3 * sb);
  row[lane + 96] = (bf16)(-f1 * sb + f3 * cb);
}

// ---------------- Flash attention (causal, GQA) ----------------
// Grid: (SEQ/64, NH, BATCH), 128 thr (4 waves). Each wave owns 16 queries.
// K (32x128) and V^T (128x32) tiles double-buffered in LDS via async copies.
// Per 32-key tile: 8 WMMA Q*K^T, online softmax (DPP max; row-sum via an
// extra WMMA against a ones matrix), 8 WMMA P*V from contiguous V^T rows.
__global__ void __launch_bounds__(128)
attn_flash(const bf16* __restrict__ q, const bf16* __restrict__ k,
           const bf16* __restrict__ v, bf16* __restrict__ y) {
  // K rows padded to 152 elems (304B == 12 banks mod 64): conflict-free
  __shared__ __align__(16) bf16 Ks[2][32][152];
  // V^T rows (one channel, 32 keys) padded to 40 elems (80B): conflict-free
  __shared__ __align__(16) bf16 Vt[2][128][40];
  __shared__ __align__(16) bf16 Ps[4][16][40];

  const int blkQ0 = blockIdx.x * 64;
  const int h = blockIdx.y, b = blockIdx.z;
  const int kvh = h / (NH / NKVH);
  const int t = threadIdx.x;
  const int wave = t >> 5, lane = t & 31;
  const int ln = lane & 15, lh = lane >> 4;
  const int qb0 = blkQ0 + wave * 16;

  const bf16* qptr  = q + (((size_t)b * NH   + h  ) * SEQ ) * HDIM + (size_t)qb0 * HDIM;
  const bf16* kptr  = k + (((size_t)b * NKVH + kvh) * SEQ ) * HDIM;
  const bf16* vtptr = v + (((size_t)b * NKVH + kvh) * HDIM) * SEQ;   // (chan, seq)

  // stage K tile (32 rows x 256B) and V^T tile (128 rows x 64B): 4 rounds each
  auto stage = [&](int buf, int key0) {
#pragma unroll
    for (int r = 0; r < 4; ++r) {
      const int c = r * 128 + t;
      { const int row = c >> 4, cb = (c & 15) * 16;
        async_load_b128(lds_addr(&Ks[buf][row][0]) + cb,
                        (const char*)kptr + (size_t)(key0 + row) * 256 + cb); }
      { const int row = c >> 2, cb = (c & 3) * 16;
        async_load_b128(lds_addr(&Vt[buf][row][0]) + cb,
                        (const char*)vtptr + (size_t)row * (SEQ * 2) + key0 * 2 + cb); }
    }
  };

  // Q fragments for K-dim 0..127 (4 x 16x32), kept in registers throughout
  union { v16bf v; uint4 u[2]; } qa[4];
#pragma unroll
  for (int c4 = 0; c4 < 4; ++c4) {
    const bf16* qrow = qptr + ln * HDIM + c4 * 32;
    qa[c4].u[0] = *(const uint4*)(qrow + 16 * lh);
    qa[c4].u[1] = *(const uint4*)(qrow + 16 * lh + 8);
  }

  // ones B-fragment for the WMMA row-sum trick
  v16bf onesf;
#pragma unroll
  for (int j = 0; j < 16; ++j) onesf[j] = (bf16)1.0f;

  v8f o[8] = {};
  float mrow[8], lrow[8], alph[8];
#pragma unroll
  for (int i = 0; i < 8; ++i) { mrow[i] = -3.0e38f; lrow[i] = 0.0f; }

  const float scale = 0.088388347648318447f;  // 1/sqrt(128)
  const int nkb = (blkQ0 + 95) >> 5;          // tiles needed by the last wave

  stage(0, 0);
  wait_async();
  __syncthreads();

  for (int kb = 0; kb < nkb; ++kb) {
    const int cur = kb & 1;
    const int key0 = kb * 32;
    if (kb + 1 < nkb) stage(cur ^ 1, (kb + 1) * 32);   // overlap with compute

    if (key0 <= qb0 + 15) {   // wave-uniform causal skip; EXEC all-ones inside
      v8f sc0 = {}, sc1 = {};
#pragma unroll
      for (int c4 = 0; c4 < 4; ++c4) {
        union { v16bf v; uint4 u[2]; } kf;
        const bf16* krow0 = &Ks[cur][ln][c4 * 32 + 16 * lh];
        kf.u[0] = *(const uint4*)krow0;
        kf.u[1] = *(const uint4*)(krow0 + 8);
        sc0 = __builtin_amdgcn_wmma_f32_16x16x32_bf16(
            false, qa[c4].v, false, kf.v, (short)0, sc0, false, false);
        const bf16* krow1 = &Ks[cur][16 + ln][c4 * 32 + 16 * lh];
        kf.u[0] = *(const uint4*)krow1;
        kf.u[1] = *(const uint4*)(krow1 + 8);
        sc1 = __builtin_amdgcn_wmma_f32_16x16x32_bf16(
            false, qa[c4].v, false, kf.v, (short)0, sc1, false, false);
      }

      // online softmax: row i+8*lh, cols across the 16 lanes of each half
#pragma unroll
      for (int i = 0; i < 8; ++i) {
        const int qrow = qb0 + i + 8 * lh;
        float s0 = (key0 + ln      <= qrow) ? sc0[i] * scale : -3.0e38f;
        float s1 = (key0 + 16 + ln <= qrow) ? sc1[i] * scale : -3.0e38f;
        const float tm = half_max16(fmaxf(s0, s1));      // pure-VALU DPP max
        const float mn = fmaxf(mrow[i], tm);
        alph[i] = __expf(mrow[i] - mn);
        mrow[i] = mn;
        const float p0 = __expf(s0 - mn);
        const float p1 = __expf(s1 - mn);
#pragma unroll
        for (int n = 0; n < 8; ++n) o[n][i] *= alph[i];
        Ps[wave][i + 8 * lh][ln]      = (bf16)p0;
        Ps[wave][i + 8 * lh][16 + ln] = (bf16)p1;
      }
      // same-wave LDS RAW: DS ops are in-order per wave; compiler inserts waits

      // P as A-fragment (16 queries x 32 keys)
      union { v16bf v; uint4 u[2]; } pa;
      pa.u[0] = *(const uint4*)(&Ps[wave][ln][16 * lh]);
      pa.u[1] = *(const uint4*)(&Ps[wave][ln][16 * lh + 8]);

      // row sums via WMMA against ones: every column of ls holds sum_k P[m,k]
      v8f ls = {};
      ls = __builtin_amdgcn_wmma_f32_16x16x32_bf16(
          false, pa.v, false, onesf, (short)0, ls, false, false);
#pragma unroll
      for (int i = 0; i < 8; ++i) lrow[i] = lrow[i] * alph[i] + ls[i];

#pragma unroll
      for (int n = 0; n < 8; ++n) {
        union { v16bf v; uint4 u[2]; } vb_;
        const bf16* vrow = &Vt[cur][n * 16 + ln][0];
        vb_.u[0] = *(const uint4*)(vrow + 16 * lh);
        vb_.u[1] = *(const uint4*)(vrow + 16 * lh + 8);
        o[n] = __builtin_amdgcn_wmma_f32_16x16x32_bf16(
            false, pa.v, false, vb_.v, (short)0, o[n], false, false);
      }
    }

    wait_async();       // next K/V tile landed while we computed
    __syncthreads();    // all waves done reading buf `cur` before it's refilled
  }

  // write y in (b*S, EMB) row-major for the output projection
#pragma unroll
  for (int n = 0; n < 8; ++n)
#pragma unroll
    for (int i = 0; i < 8; ++i) {
      const int qrow = qb0 + i + 8 * lh;
      const float val = o[n][i] / lrow[i];
      y[((size_t)b * SEQ + qrow) * EMB + h * HDIM + n * 16 + ln] = (bf16)val;
    }
}

// ---------------- launch ----------------
extern "C" void kernel_launch(void* const* d_in, const int* in_sizes, int n_in,
                              void* d_out, int out_size, void* d_ws, size_t ws_size,
                              hipStream_t stream) {
  const float* x  = (const float*)d_in[0];
  const float* Wq = (const float*)d_in[1];
  const float* Wk = (const float*)d_in[2];
  const float* Wv = (const float*)d_in[3];
  const float* Wp = (const float*)d_in[4];
  const float* qg = (const float*)d_in[5];

  // workspace layout (bf16 elems). total = 38 Mi elems = 76 MiB.
  bf16* ws = (bf16*)d_ws;
  const size_t M1 = 1024 * 1024;
  bf16* xb  = ws;             // x              : 8 Mi
  bf16* wqb = xb  + 8 * M1;   // Wq             : 4 Mi
  bf16* wkb = wqb + 4 * M1;   // Wk             : 1 Mi
  bf16* wvb = wkb + 1 * M1;   // Wv             : 1 Mi
  bf16* wpb = wvb + 1 * M1;   // Wp             : 4 Mi
  bf16* qb  = wpb + 4 * M1;   // q  (b,h,s,e)   : 8 Mi
  bf16* kb  = qb  + 8 * M1;   // k  (b,kv,s,e)  : 2 Mi
  bf16* vb  = kb  + 2 * M1;   // v^T (b,kv,e,s) : 2 Mi
  bf16* yb  = vb  + 2 * M1;   // attn out       : 8 Mi

  auto cvt = [&](const float* src, bf16* dst, int n) {
    const int n4 = n >> 2;
    cvt_f32_bf16<<<(n4 + 255) / 256, 256, 0, stream>>>(src, dst, n4);
  };
  cvt(x,  xb,  BATCH * SEQ * EMB);
  cvt(Wq, wqb, EMB * EMB);
  cvt(Wk, wkb, KVD * EMB);
  cvt(Wv, wvb, KVD * EMB);
  cvt(Wp, wpb, EMB * EMB);

  // QKV projections; V is written head-transposed for the attention kernel
  gemm_bf16_wmma<bf16, 1><<<dim3(EMB / 128, MR / 128), 256, 0, stream>>>(
      xb, wqb, qb, MR, EMB, EMB);
  gemm_bf16_wmma<bf16, 1><<<dim3(KVD / 128, MR / 128), 256, 0, stream>>>(
      xb, wkb, kb, MR, KVD, EMB);
  gemm_bf16_wmma<bf16, 2><<<dim3(KVD / 128, MR / 128), 256, 0, stream>>>(
      xb, wvb, vb, MR, KVD, EMB);

  rmsnorm_rope<<<(BATCH * NH * SEQ) / 8, 256, 0, stream>>>(
      qb, qg, BATCH * NH * SEQ, NH);
  rmsnorm_rope<<<(BATCH * NKVH * SEQ) / 8, 256, 0, stream>>>(
      kb, nullptr, BATCH * NKVH * SEQ, NKVH);

  attn_flash<<<dim3(SEQ / 64, NH, BATCH), 128, 0, stream>>>(qb, kb, vb, yb);

  // output projection -> fp32 d_out
  gemm_bf16_wmma<float, 0><<<dim3(EMB / 128, MR / 128), 256, 0, stream>>>(
      yb, wpb, (float*)d_out, MR, EMB, EMB);
}
